// cross_AttnBlock_26225070309677
// MI455X (gfx1250) — matrix operand order repaired and checked by reading, not compiled, not run on hardware
//
#include <hip/hip_runtime.h>
#include <hip/hip_bf16.h>

typedef __bf16 bf16;
typedef __attribute__((ext_vector_type(16))) __bf16 v16bf;
typedef __attribute__((ext_vector_type(8)))  __bf16 bf16x8;
typedef __attribute__((ext_vector_type(8)))  float  v8f;

#define C_DIM 256
#define N_SP  4096
#define B_SZ  8

#if __has_builtin(__builtin_amdgcn_tensor_load_to_lds)
#define HAS_TDM 1
#else
#define HAS_TDM 0
#endif

#if __has_builtin(__builtin_amdgcn_s_wait_tensorcnt)
#define WAIT_TCNT(n) __builtin_amdgcn_s_wait_tensorcnt(n)
#else
#define WAIT_TCNT(n) asm volatile("s_wait_tensorcnt %0" :: "i"(n) : "memory")
#endif

// ---------- fragment helpers (contiguous k order k = 16h + s, same in A and B) ----------

__device__ __forceinline__ v16bf frag_cat(bf16x8 lo, bf16x8 hi) {
  return __builtin_shufflevector(lo, hi, 0,1,2,3,4,5,6,7,8,9,10,11,12,13,14,15);
}

// A-matrix 16x32: per-lane halfs = 16 contiguous K values (p pre-offset by 16*halfsel)
__device__ __forceinline__ v16bf load_a_frag(const bf16* p) {
  bf16x8 lo = *(const bf16x8*)(p);
  bf16x8 hi = *(const bf16x8*)(p + 8);
  return frag_cat(lo, hi);
}

// B-matrix 32x16: per-lane halfs = 16 contiguous K values (p pre-offset by 16*halfsel)
__device__ __forceinline__ v16bf load_b_frag(const bf16* p) {
  bf16x8 lo = *(const bf16x8*)(p);
  bf16x8 hi = *(const bf16x8*)(p + 8);
  return frag_cat(lo, hi);
}

__device__ __forceinline__ v8f wmma_bf16(v16bf a, v16bf b, v8f c) {
  return __builtin_amdgcn_wmma_f32_16x16x32_bf16(false, a, false, b, (short)0, c, false, false);
}

#if HAS_TDM
// ---------- Tensor Data Mover: 2D tile load, bf16 elements, optional LDS row pad ----
typedef __attribute__((ext_vector_type(4))) unsigned int v4u;
typedef __attribute__((ext_vector_type(8))) int v8i;
typedef __attribute__((ext_vector_type(4))) int v4i;

__device__ __forceinline__ unsigned lds_off(const void* p) {
  return (unsigned)(unsigned long long)(uintptr_t)p;   // addr[31:0] = LDS offset
}

// pad_interval_code: interval = 2^(code+1) dwords; pad_amount in dwords (0 = no pad)
__device__ __forceinline__ void tdm_load_2d(
    unsigned lds_addr, const void* gaddr,
    unsigned tensor_d0, unsigned tensor_d1, unsigned long long stride0,
    unsigned tile_d0, unsigned tile_d1,
    unsigned pad_interval_code, unsigned pad_amount_dw)
{
  unsigned long long ga = (unsigned long long)(uintptr_t)gaddr;
  v4u g0;
  g0[0] = 1u;                                        // count=1, user mode
  g0[1] = lds_addr;                                  // D#.lds_addr
  g0[2] = (unsigned)ga;                              // global_addr[31:0]
  g0[3] = (unsigned)((ga >> 32) & 0x1FFFFFFull) | (2u << 30);   // [56:32] | type=2
  unsigned w0 = (1u << 16);                          // data_size = 2 bytes
  if (pad_amount_dw)
    w0 |= (1u << 20) | (pad_interval_code << 22) | ((pad_amount_dw - 1u) << 25);
  v8i g1;
  g1[0] = (int)w0;
  g1[1] = (int)((tensor_d0 & 0xFFFFu) << 16);
  g1[2] = (int)(((tensor_d0 >> 16) & 0xFFFFu) | ((tensor_d1 & 0xFFFFu) << 16));
  g1[3] = (int)(((tensor_d1 >> 16) & 0xFFFFu) | ((tile_d0 & 0xFFFFu) << 16));
  g1[4] = (int)(tile_d1 & 0xFFFFu);                  // tile_dim1, tile_dim2=0
  g1[5] = (int)(unsigned)(stride0 & 0xFFFFFFFFull);  // tensor_dim0_stride lo
  g1[6] = (int)(unsigned)((stride0 >> 32) & 0xFFFFull);
  g1[7] = 0;
  v4i g2 = {1, 1, 0, 0};                             // tensor_dim2/3 = 1, tile_dim3 = 0
  v4i g3 = {0, 0, 0, 0};
#if defined(__clang_major__) && __clang_major__ >= 23
  v8i g4 = {0, 0, 0, 0, 0, 0, 0, 0};
  __builtin_amdgcn_tensor_load_to_lds(g0, g1, g2, g3, g4, 0);
#else
  __builtin_amdgcn_tensor_load_to_lds(g0, g1, g2, g3, 0);
#endif
}
#endif  // HAS_TDM

// ---------- kernel 0: fp32 -> bf16 weight conversion ----------

__global__ __launch_bounds__(256) void prep_weights(
    const float* __restrict__ qw, const float* __restrict__ kw,
    const float* __restrict__ vw, const float* __restrict__ pw,
    bf16* __restrict__ qo, bf16* __restrict__ ko,
    bf16* __restrict__ vo, bf16* __restrict__ po)
{
  int i = blockIdx.x * 256 + threadIdx.x;   // 65536 elements each
  qo[i] = (bf16)qw[i];
  ko[i] = (bf16)kw[i];
  vo[i] = (bf16)vw[i];
  po[i] = (bf16)pw[i];
}

// ---------- kernel 1: groupnorm stats: mean/rstd per (b, x|y, group) ----------

__global__ __launch_bounds__(256) void gn_stats(const float* __restrict__ xy,
                                                float* __restrict__ stats)
{
  __shared__ float s1[256], s2[256];
  int bid = blockIdx.x;            // 256 = (b*2+which)*16 + g
  int bw  = bid >> 4;
  int g   = bid & 15;
  int b   = bw >> 1, which = bw & 1;
  const float* base = xy + ((long long)b * 2 * C_DIM + which * C_DIM + g * 16) * N_SP;
  float acc1 = 0.f, acc2 = 0.f;
  for (int i = threadIdx.x; i < 16 * N_SP; i += 256) {
    int cl = i >> 12, n = i & (N_SP - 1);
    float v = base[(long long)cl * N_SP + n];
    acc1 += v; acc2 += v * v;
  }
  s1[threadIdx.x] = acc1; s2[threadIdx.x] = acc2;
  __syncthreads();
  for (int s = 128; s > 0; s >>= 1) {
    if (threadIdx.x < s) { s1[threadIdx.x] += s1[threadIdx.x + s]; s2[threadIdx.x] += s2[threadIdx.x + s]; }
    __syncthreads();
  }
  if (threadIdx.x == 0) {
    float inv_n = 1.0f / (16.0f * N_SP);
    float mean = s1[0] * inv_n;
    float var  = s2[0] * inv_n - mean * mean;
    stats[bid * 2 + 0] = mean;
    stats[bid * 2 + 1] = rsqrtf(var + 1e-6f);
  }
}

// ---------- kernel 2: normalize + transpose (B,C,N) f32 -> (B,N,C) bf16 ----------

__global__ __launch_bounds__(256) void gn_apply(
    const float* __restrict__ xy, const float* __restrict__ stats,
    const float* __restrict__ nw, const float* __restrict__ nb,
    bf16* __restrict__ xn_t, bf16* __restrict__ yn_t)
{
  __shared__ float tile[64][65];
  int n0 = blockIdx.x * 64;
  int c0 = blockIdx.y * 64;
  int bw = blockIdx.z;
  int b = bw >> 1, which = bw & 1;
  const float* src = xy + ((long long)b * 2 * C_DIM + which * C_DIM) * N_SP;
  for (int i = threadIdx.x; i < 64 * 64; i += 256) {
    int cl = i >> 6, nl = i & 63;
    int c = c0 + cl;
    int g = c >> 4;
    float mean = stats[(bw * 16 + g) * 2 + 0];
    float rstd = stats[(bw * 16 + g) * 2 + 1];
    float v = src[(long long)c * N_SP + n0 + nl];
    tile[cl][nl] = (v - mean) * rstd * nw[c] + nb[c];
  }
  __syncthreads();
  bf16* dst = (which ? yn_t : xn_t) + (long long)b * N_SP * C_DIM;
  for (int i = threadIdx.x; i < 64 * 64; i += 256) {
    int nl = i >> 6, cl = i & 63;
    dst[(long long)(n0 + nl) * C_DIM + c0 + cl] = (bf16)tile[cl][nl];
  }
}

// ---------- generic WMMA GEMM:  D[m,n] = sum_c A[m,c]*Bt[n,c] (+bias)(+resid) ----------

template<bool OUTF32, bool BIAS_ROW, bool RES>
__global__ __launch_bounds__(256) void gemm_abT(
    const bf16* __restrict__ A, const bf16* __restrict__ Bt, void* __restrict__ Dv,
    const float* __restrict__ bias, const float* __restrict__ resid,
    long long sA, long long sB, long long sD, long long sR, int ldd)
{
  int batch = blockIdx.z;
  const bf16* Ab  = A  + batch * sA;
  const bf16* Btb = Bt + batch * sB;
  int tid = threadIdx.x, wave = tid >> 5, lane = tid & 31;
  int l15 = lane & 15, hsel = (lane >> 4) & 1;
  int wm = wave >> 2, wn = wave & 3;
  int m0 = blockIdx.x * 32 + wm * 16;
  int n0 = blockIdx.y * 256 + wn * 64;

  v8f acc[4] = { {}, {}, {}, {} };
  const bf16* arow = Ab + (long long)(m0 + l15) * C_DIM;
#pragma unroll
  for (int kk = 0; kk < 8; ++kk) {
    int cb = kk * 32;
    v16bf a = load_a_frag(arow + cb + 16 * hsel);
    int kb = cb + 16 * hsel;
#pragma unroll
    for (int j = 0; j < 4; ++j) {
      v16bf bfrag = load_b_frag(Btb + (long long)(n0 + j * 16 + l15) * C_DIM + kb);
      acc[j] = wmma_bf16(a, bfrag, acc[j]);
    }
  }

  float* Df = (float*)Dv;
  bf16*  Db = (bf16*)Dv;
#pragma unroll
  for (int j = 0; j < 4; ++j) {
    int col = n0 + j * 16 + l15;
#pragma unroll
    for (int r = 0; r < 8; ++r) {
      int row = m0 + r + 8 * hsel;
      float val = acc[j][r] + (BIAS_ROW ? bias[row] : bias[col]);
      long long di = batch * sD + (long long)row * ldd + col;
      if (RES) val += resid[batch * sR + (long long)row * ldd + col];
      if (OUTF32) Df[di] = val; else Db[di] = (bf16)val;
    }
  }
}

// ---------- fused flash-attention kernel ----------
// grid: 512 = B * (N/64); block = 128 thr = 4 waves (16 queries/wave).
// q_t,k_t: (B,N,C) bf16; v: (B,C,N) bf16; h_t: (B,N,C) bf16.
// K/V tiles double-buffered in LDS; TDM streams tile kt+1 while computing kt.

__global__ __launch_bounds__(128) void attn_flash(
    const bf16* __restrict__ q_t, const bf16* __restrict__ k_t,
    const bf16* __restrict__ v,   bf16* __restrict__ h_t)
{
  __shared__ bf16 kTile[2][64][264];   // keys x channels (+16B row pad)
  __shared__ bf16 vTile[2][256][72];   // channels x keys (+16B row pad)
  __shared__ bf16 pbuf[4][16][72];     // per-wave P relayout buffer

  int b  = blockIdx.x >> 6;
  int q0 = (blockIdx.x & 63) * 64;
  int tid = threadIdx.x, wave = tid >> 5, lane = tid & 31;
  int l15 = lane & 15, hsel = (lane >> 4) & 1;
  int qw0 = q0 + wave * 16;

  const bf16* qb = q_t + (long long)b * N_SP * C_DIM;
  const bf16* kb = k_t + (long long)b * N_SP * C_DIM;
  const bf16* vb = v   + (long long)b * C_DIM * N_SP;
  bf16*       hb = h_t + (long long)b * N_SP * C_DIM;

  // resident Q fragments: 16 queries x 256 channels
  v16bf qf[8];
  {
    const bf16* qrow = qb + (long long)(qw0 + l15) * C_DIM;
#pragma unroll
    for (int kk = 0; kk < 8; ++kk)
      qf[kk] = load_a_frag(qrow + kk * 32 + 16 * hsel);
  }

  v8f  oacc[16];
#pragma unroll
  for (int ct = 0; ct < 16; ++ct) oacc[ct] = (v8f){};
  float m_r[8], l_r[8];
#pragma unroll
  for (int r = 0; r < 8; ++r) { m_r[r] = -3.0e38f; l_r[r] = 0.f; }

  const float scale = 0.0625f;  // 256^-0.5

#if HAS_TDM
  if (wave == 0) {  // prologue: DMA tile 0 (K row: 512B -> pad code 6; V row: 128B -> code 4)
    tdm_load_2d(lds_off(&kTile[0][0][0]), kb, C_DIM, N_SP, C_DIM, C_DIM, 64, 6, 4);
    tdm_load_2d(lds_off(&vTile[0][0][0]), vb, N_SP, C_DIM, N_SP, 64, C_DIM, 4, 4);
  }
#endif

#pragma unroll 1
  for (int kt = 0; kt < 64; ++kt) {
    int k0 = kt * 64;
    int buf = kt & 1;

#if HAS_TDM
    if (wave == 0) {
      if (kt + 1 < 64) {   // stream next tile into the other buffer
        int nb = buf ^ 1, nk0 = k0 + 64;
        tdm_load_2d(lds_off(&kTile[nb][0][0]), kb + (long long)nk0 * C_DIM,
                    C_DIM, N_SP, C_DIM, C_DIM, 64, 6, 4);
        tdm_load_2d(lds_off(&vTile[nb][0][0]), vb + nk0,
                    N_SP, C_DIM, N_SP, 64, C_DIM, 4, 4);
        WAIT_TCNT(2);      // current tile's pair complete; next pair in flight
      } else {
        WAIT_TCNT(0);
      }
    }
    __syncthreads();
#else
    __syncthreads();       // previous readers done with this buffer
    for (int i = tid; i < 2048; i += 128) {
      int r = i >> 5, s = i & 31;
      *(bf16x8*)&kTile[buf][r][s * 8] = *(const bf16x8*)(kb + (long long)(k0 + r) * C_DIM + s * 8);
    }
    for (int i = tid; i < 2048; i += 128) {
      int r = i >> 3, s = i & 7;
      *(bf16x8*)&vTile[buf][r][s * 8] = *(const bf16x8*)(vb + (long long)r * N_SP + k0 + s * 8);
    }
    __syncthreads();
    if (kt + 1 < 64) {
      __builtin_prefetch(kb + (long long)(k0 + 64) * C_DIM + tid * 128, 0, 1);
      __builtin_prefetch(vb + (long long)(tid * 2) * N_SP + k0 + 64, 0, 1);
    }
#endif

    // ---- S = (Q Kt) * scale : 16 queries x 64 keys (SW-pipelined B frags) ----
    v8f s_acc[4] = { {}, {}, {}, {} };
    {
      v16bf bcur[4], bnxt[4];
      const bf16* kbase = &kTile[buf][0][0];
#pragma unroll
      for (int nt = 0; nt < 4; ++nt)
        bcur[nt] = load_b_frag(kbase + (nt * 16 + l15) * 264 + 16 * hsel);
#pragma unroll
      for (int kk = 0; kk < 8; ++kk) {
        if (kk < 7) {
          int coff = (kk + 1) * 32 + 16 * hsel;
#pragma unroll
          for (int nt = 0; nt < 4; ++nt)
            bnxt[nt] = load_b_frag(kbase + (nt * 16 + l15) * 264 + coff);
        }
#pragma unroll
        for (int nt = 0; nt < 4; ++nt)
          s_acc[nt] = wmma_bf16(qf[kk], bcur[nt], s_acc[nt]);
#pragma unroll
        for (int nt = 0; nt < 4; ++nt) bcur[nt] = bnxt[nt];
      }
    }
#pragma unroll
    for (int nt = 0; nt < 4; ++nt)
#pragma unroll
      for (int r = 0; r < 8; ++r) s_acc[nt][r] *= scale;

    // ---- online softmax (row stats within each 16-lane half-group) ----
    float rmax[8], rsum[8], mnew[8], corr[8];
#pragma unroll
    for (int r = 0; r < 8; ++r) {
      float mx = s_acc[0][r];
      mx = fmaxf(mx, s_acc[1][r]); mx = fmaxf(mx, s_acc[2][r]); mx = fmaxf(mx, s_acc[3][r]);
      rmax[r] = mx;
    }
#pragma unroll
    for (int r = 0; r < 8; ++r) {
#pragma unroll
      for (int off = 8; off > 0; off >>= 1)
        rmax[r] = fmaxf(rmax[r], __shfl_xor(rmax[r], off, 16));
      mnew[r] = fmaxf(m_r[r], rmax[r]);
      corr[r] = __expf(m_r[r] - mnew[r]);
      m_r[r]  = mnew[r];
      rsum[r] = 0.f;
    }
#pragma unroll
    for (int nt = 0; nt < 4; ++nt)
#pragma unroll
      for (int r = 0; r < 8; ++r) {
        float p = __expf(s_acc[nt][r] - mnew[r]);
        s_acc[nt][r] = p;
        rsum[r] += p;
      }
#pragma unroll
    for (int r = 0; r < 8; ++r) {
#pragma unroll
      for (int off = 8; off > 0; off >>= 1)
        rsum[r] += __shfl_xor(rsum[r], off, 16);
      l_r[r] = l_r[r] * corr[r] + rsum[r];
    }
#pragma unroll
    for (int ct = 0; ct < 16; ++ct)
#pragma unroll
      for (int r = 0; r < 8; ++r) oacc[ct][r] *= corr[r];

    // ---- relayout P (C-layout) -> A-fragments via per-wave LDS ----
#pragma unroll
    for (int nt = 0; nt < 4; ++nt)
#pragma unroll
      for (int r = 0; r < 8; ++r)
        pbuf[wave][r + 8 * hsel][nt * 16 + l15] = (bf16)s_acc[nt][r];
    asm volatile("s_wait_dscnt 0x0" ::: "memory");

    // ---- O += P * Vt : 16 queries x 256 channels (SW-pipelined B frags) ----
    {
      const bf16* vbase = &vTile[buf][0][0];
#pragma unroll
      for (int kk2 = 0; kk2 < 2; ++kk2) {
        v16bf pfrag = load_a_frag(&pbuf[wave][l15][kk2 * 32 + 16 * hsel]);
        int koff = kk2 * 32 + 16 * hsel;
        v16bf bcur = load_b_frag(vbase + l15 * 72 + koff);
#pragma unroll
        for (int ct = 0; ct < 16; ++ct) {
          v16bf bnxt;
          if (ct < 15) bnxt = load_b_frag(vbase + ((ct + 1) * 16 + l15) * 72 + koff);
          oacc[ct] = wmma_bf16(pfrag, bcur, oacc[ct]);
          bcur = bnxt;
        }
      }
    }
    __syncthreads();   // all waves done with this buffer before TDM refills it
  }

  // ---- epilogue: normalize and store h_t (B,N,C) bf16 ----
#pragma unroll
  for (int r = 0; r < 8; ++r) {
    float inv = 1.0f / l_r[r];
    int row = qw0 + r + 8 * hsel;
#pragma unroll
    for (int ct = 0; ct < 16; ++ct)
      hb[(long long)row * C_DIM + ct * 16 + l15] = (bf16)(oacc[ct][r] * inv);
  }
}

// ---------- host launch ----------

extern "C" void kernel_launch(void* const* d_in, const int* in_sizes, int n_in,
                              void* d_out, int out_size, void* d_ws, size_t ws_size,
                              hipStream_t stream)
{
  (void)in_sizes; (void)n_in; (void)out_size; (void)ws_size;
  const float* xy     = (const float*)d_in[0];
  const float* norm_w = (const float*)d_in[1];
  const float* norm_b = (const float*)d_in[2];
  const float* q_w    = (const float*)d_in[3];
  const float* q_b    = (const float*)d_in[4];
  const float* k_w    = (const float*)d_in[5];
  const float* k_b    = (const float*)d_in[6];
  const float* v_w    = (const float*)d_in[7];
  const float* v_b    = (const float*)d_in[8];
  const float* p_w    = (const float*)d_in[9];
  const float* p_b    = (const float*)d_in[10];

  char* ws = (char*)d_ws;
  size_t off = 0;
  auto take = [&](size_t bytes) -> void* {
    void* p = ws + off;
    off = (off + bytes + 255) & ~(size_t)255;
    return p;
  };
  const size_t W_ELEMS = (size_t)C_DIM * C_DIM;
  const size_t T_ELEMS = (size_t)B_SZ * N_SP * C_DIM;
  bf16* qw_bf = (bf16*)take(W_ELEMS * 2);
  bf16* kw_bf = (bf16*)take(W_ELEMS * 2);
  bf16* vw_bf = (bf16*)take(W_ELEMS * 2);
  bf16* pw_bf = (bf16*)take(W_ELEMS * 2);
  float* stats = (float*)take(256 * 2 * sizeof(float));
  bf16* xn_t = (bf16*)take(T_ELEMS * 2);
  bf16* yn_t = (bf16*)take(T_ELEMS * 2);
  bf16* q_t  = (bf16*)take(T_ELEMS * 2);
  bf16* k_t  = (bf16*)take(T_ELEMS * 2);
  bf16* vmat = (bf16*)take(T_ELEMS * 2);
  bf16* h_t  = (bf16*)take(T_ELEMS * 2);

  const long long sT = (long long)N_SP * C_DIM;

  prep_weights<<<256, 256, 0, stream>>>(q_w, k_w, v_w, p_w, qw_bf, kw_bf, vw_bf, pw_bf);
  gn_stats<<<256, 256, 0, stream>>>(xy, stats);
  gn_apply<<<dim3(64, 4, 16), 256, 0, stream>>>(xy, stats, norm_w, norm_b, xn_t, yn_t);

  // q_t[n,o] = sum_c yn_t[n,c] * q_w[o,c] + q_b[o]
  gemm_abT<false, false, false><<<dim3(128, 1, B_SZ), 256, 0, stream>>>(
      yn_t, qw_bf, q_t, q_b, nullptr, sT, 0, sT, 0, C_DIM);
  // k_t[n,o] = sum_c xn_t[n,c] * k_w[o,c] + k_b[o]
  gemm_abT<false, false, false><<<dim3(128, 1, B_SZ), 256, 0, stream>>>(
      xn_t, kw_bf, k_t, k_b, nullptr, sT, 0, sT, 0, C_DIM);
  // v[o,n] = sum_c v_w[o,c] * xn_t[n,c] + v_b[o]
  gemm_abT<false, true, false><<<dim3(8, 16, B_SZ), 256, 0, stream>>>(
      vw_bf, xn_t, vmat, v_b, nullptr, 0, sT, sT, 0, N_SP);

  attn_flash<<<512, 128, 0, stream>>>(q_t, k_t, vmat, h_t);

  // out[o,n] = x[o,n] + sum_c proj_w[o,c] * h_t[n,c] + proj_b[o]
  gemm_abT<true, true, true><<<dim3(8, 16, B_SZ), 256, 0, stream>>>(
      pw_bf, h_t, d_out, p_b, xy, 0, sT, sT, (long long)2 * C_DIM * N_SP, N_SP);
}
